// HGT_18013092839761
// MI455X (gfx1250) — compile-verified, matrix operand order checked by reading, not run on hardware
//
#include <hip/hip_runtime.h>
#include <math.h>

// ---------------------------------------------------------------------------
// Types for CDNA5 WMMA
// ---------------------------------------------------------------------------
typedef __bf16 bh16;
typedef __attribute__((ext_vector_type(16))) __bf16 v16bf;
typedef __attribute__((ext_vector_type(8)))  float  v8f;
typedef int v4i_ __attribute__((vector_size(16)));   // matches builtin param type

#define LDSS 40   // padded LDS row stride in bf16 elems: 80B rows -> 16B aligned chunks,
                  // conflict-free 16-lane fragment reads (20-word stride mod 64)

// CDNA5 async global->LDS path (guarded: falls back to vector copy if the
// toolchain does not expose the gfx1250 async builtins)
#if defined(__AMDGCN__) && __has_builtin(__builtin_amdgcn_global_load_async_to_lds_b128) && __has_builtin(__builtin_amdgcn_s_wait_asynccnt)
#define USE_ASYNC_LDS 1
#else
#define USE_ASYNC_LDS 0
#endif

__device__ __forceinline__ void copy16_g2l(const bh16* g, bh16* l)
{
#if USE_ASYNC_LDS
    __builtin_amdgcn_global_load_async_to_lds_b128(
        (__attribute__((address_space(1))) v4i_*)g,
        (__attribute__((address_space(3))) v4i_*)l, 0, 0);
#else
    *(uint4*)l = *(const uint4*)g;
#endif
}

__device__ __forceinline__ void wait_async_lds()
{
#if USE_ASYNC_LDS
    __builtin_amdgcn_s_wait_asynccnt(0);
#endif
}

// ---- stage one 128x32 tile into LDS.  Out-of-range rows are CLAMPED to the
//      last valid row instead of guarded: tile row r only ever contributes to
//      C row m0+r, which is never stored when >= M, so duplicated data is
//      harmless and staging stays branch-free (no exec-mask chains). ----
// bf16 source (row stride ld elems): 512 x 16B chunks, 2 per thread
__device__ __forceinline__ void stage_tile(const bh16* __restrict__ g, long ld,
                                           int mrem, bh16* lds, int tid)
{
#pragma unroll
    for (int i = 0; i < 2; ++i) {
        int idx = tid + i * 256;
        int r = idx >> 2, c = (idx & 3) << 3;
        int rr = r < mrem ? r : mrem - 1;
        copy16_g2l(g + (long)rr * ld + c, lds + r * LDSS + c);
    }
}
// fp32 source: 1024 x 16B chunks (4 floats), convert + packed 8B LDS store
__device__ __forceinline__ void stage_tile(const float* __restrict__ g, long ld,
                                           int mrem, bh16* lds, int tid)
{
    float4 v[4];
#pragma unroll
    for (int i = 0; i < 4; ++i) {           // batch the 4 global loads
        int idx = tid + i * 256;
        int r = idx >> 3, c = (idx & 7) << 2;
        int rr = r < mrem ? r : mrem - 1;
        v[i] = *(const float4*)(g + (long)rr * ld + c);
    }
#pragma unroll
    for (int i = 0; i < 4; ++i) {           // convert + store
        int idx = tid + i * 256;
        int r = idx >> 3, c = (idx & 7) << 2;
        union { bh16 h[4]; uint2 u; } pk;
        pk.h[0] = (bh16)v[i].x; pk.h[1] = (bh16)v[i].y;
        pk.h[2] = (bh16)v[i].z; pk.h[3] = (bh16)v[i].w;
        *(uint2*)(lds + r * LDSS + c) = pk.u;
    }
}

// ---------------------------------------------------------------------------
// Tiled GEMM:  C[M,N] = act( A[M,K] * B[K,N] + bias[N] )
// A: AT (bf16/f32) row-major.  Bt: bf16, PRE-TRANSPOSED [N][K] row-major.
// Block = 256 threads (8 wave32 waves), tile 128x128, K-step 32,
// double-buffered LDS with async global->LDS prefetch of the next K-tile.
// Each wave computes a 32x64 sub-tile = 2x4 WMMA 16x16x32 bf16 accumulators.
// Requires N % 128 == 0, K % 32 == 0.  act: 0 none, 1 leaky_relu(0.01)
// ---------------------------------------------------------------------------
template <typename AT, typename CT>
__global__ __launch_bounds__(256) void gemm_wmma(
    const AT* __restrict__ A, const bh16* __restrict__ Bt,
    const float* __restrict__ bias, CT* __restrict__ C,
    int M, int N, int K, int act)
{
    __shared__ bh16 lA[2][128 * LDSS];
    __shared__ bh16 lB[2][128 * LDSS];

    const int tid  = threadIdx.x;
    const int lane = tid & 31;
    const int wave = tid >> 5;
    const int wm   = wave >> 1;   // 0..3 : 32-row strip
    const int wn   = wave & 1;    // 0..1 : 64-col strip
    const int m0   = blockIdx.x * 128;
    const int n0   = blockIdx.y * 128;
    const int lr   = lane & 15;
    const int kg   = lane >> 4;
    const int mrem = M - m0;

    v8f acc[2][4];
    for (int i = 0; i < 2; ++i)
        for (int j = 0; j < 4; ++j)
            for (int r = 0; r < 8; ++r) acc[i][j][r] = 0.0f;

    const int T = K >> 5;
    // prologue: stage tile 0
    stage_tile(A + (long)m0 * K, (long)K, mrem, lA[0], tid);
    stage_tile(Bt + (long)n0 * K, (long)K, 128, lB[0], tid);

    for (int t = 0; t < T; ++t) {
        wait_async_lds();
        __syncthreads();
        if (t + 1 < T) {   // prefetch next K-tile into the other buffer
            int kk = (t + 1) << 5;
            stage_tile(A + (long)m0 * K + kk, (long)K, mrem, lA[(t + 1) & 1], tid);
            stage_tile(Bt + (long)n0 * K + kk, (long)K, 128, lB[(t + 1) & 1], tid);
        }
        const bh16* bufA = lA[t & 1];
        const bh16* bufB = lB[t & 1];

        // fragments: two 16B LDS reads each (CDNA5 16-bit A/B VGPR layout)
        union F { v16bf v; uint4 q[2]; };
        F a[2], b[4];
#pragma unroll
        for (int mt = 0; mt < 2; ++mt) {
            const char* p = (const char*)&bufA[(wm * 32 + mt * 16 + lr) * LDSS];
            a[mt].q[0] = *(const uint4*)(p + kg * 16);
            a[mt].q[1] = *(const uint4*)(p + 32 + kg * 16);
        }
#pragma unroll
        for (int nt = 0; nt < 4; ++nt) {
            const char* p = (const char*)&bufB[(wn * 64 + nt * 16 + lr) * LDSS];
            b[nt].q[0] = *(const uint4*)(p + kg * 16);
            b[nt].q[1] = *(const uint4*)(p + 32 + kg * 16);
        }
#pragma unroll
        for (int mt = 0; mt < 2; ++mt)
#pragma unroll
            for (int nt = 0; nt < 4; ++nt)
                acc[mt][nt] = __builtin_amdgcn_wmma_f32_16x16x32_bf16(
                    false, a[mt].v, false, b[nt].v,
                    (short)0, acc[mt][nt], false, false);
    }

    // epilogue: C/D layout — col = lane&15, row = vgpr + 8*(lane>>4)
    for (int mt = 0; mt < 2; ++mt) {
        for (int nt = 0; nt < 4; ++nt) {
            int col = n0 + wn * 64 + nt * 16 + lr;
#pragma unroll
            for (int r = 0; r < 8; ++r) {
                int row = m0 + wm * 32 + mt * 16 + kg * 8 + r;
                if (row < M) {
                    float v = acc[mt][nt][r] + bias[col];
                    if (act == 1) v = v > 0.0f ? v : 0.01f * v;
                    C[(long)row * N + col] = (CT)v;
                }
            }
        }
    }
}

// ---------------------------------------------------------------------------
// Weight pre-transpose: W[K][N] f32 -> Wt[N][K] bf16  (write-coalesced)
// ---------------------------------------------------------------------------
__global__ void transpose_to_bf16(const float* __restrict__ W, bh16* __restrict__ Wt,
                                  int K, int N)
{
    long i = (long)blockIdx.x * 256 + threadIdx.x;
    if (i >= (long)K * N) return;
    long n = i / K, k = i % K;
    Wt[n * K + k] = (bh16)W[k * N + n];
}

// ---------------------------------------------------------------------------
// Fold per-head relation transform into projection weights, output TRANSPOSED bf16:
//   WfT[h*32+e, i] = sum_d W[i, h*32+d] * rel[h,d,e]   (i < 256)
//   bf[h*32+e]     = sum_d bias[h*32+d] * rel[h,d,e]   (block 256 == bias row)
// grid = 257 blocks x 256 threads
// ---------------------------------------------------------------------------
__global__ void fold_rel(const float* __restrict__ W, const float* __restrict__ bias,
                         const float* __restrict__ rel,
                         bh16* __restrict__ WfT, float* __restrict__ bf)
{
    int c = threadIdx.x;             // output column 0..255
    int i = blockIdx.x;              // row, 256 == bias
    int h = c >> 5;
    int e2 = c & 31;
    const float* r = rel + h * 1024 + e2;       // rel[h][d][e2], stride 32 over d
    const float* w = (i < 256) ? (W + (long)i * 256 + h * 32) : (bias + h * 32);
    float s = 0.0f;
#pragma unroll 8
    for (int d = 0; d < 32; ++d) s += w[d] * r[d * 32];
    if (i < 256) WfT[(long)c * 256 + i] = (bh16)s;
    else         bf[c] = s;
}

// ---------------------------------------------------------------------------
// Helpers
// ---------------------------------------------------------------------------
__global__ void fill_kernel(float* __restrict__ p, float v, long n)
{
    long i = (long)blockIdx.x * 256 + threadIdx.x;
    if (i < n) p[i] = v;
}

__device__ __forceinline__ void atomicMaxF(float* addr, float val)
{
    unsigned* a = (unsigned*)addr;
    unsigned old = *a;
    while (true) {
        if (__uint_as_float(old) >= val) break;
        unsigned assumed = old;
        old = atomicCAS(a, assumed, __float_as_uint(val));
        if (old == assumed) break;
    }
}

// wave-per-edge: logits[e,h] = p_rel[h]*invsqrtD * dot32(k[src,h,:], q[dst,h,:])
__global__ __launch_bounds__(256) void edge_logits(
    const int* __restrict__ src, const int* __restrict__ dst,
    const bh16* __restrict__ Kf, const bh16* __restrict__ Qf,
    const float* __restrict__ prel, float* __restrict__ logits,
    float* __restrict__ rmax, int E, float scale)
{
    long gid = (long)blockIdx.x * 256 + threadIdx.x;
    long edge = gid >> 5;
    int lane = (int)(gid & 31);
    if (edge >= E) return;
    long row = src[edge], col = dst[edge];
    const bh16* kp = Kf + row * 256;
    const bh16* qp = Qf + col * 256;
#pragma unroll
    for (int h = 0; h < 8; ++h) {
        float p = (float)kp[h * 32 + lane] * (float)qp[h * 32 + lane];
        for (int m = 16; m >= 1; m >>= 1) p += __shfl_xor(p, m, 32);
        if (lane == 0) {
            float lg = p * prel[h] * scale;
            logits[edge * 8 + h] = lg;
            atomicMaxF(&rmax[col * 8 + h], lg);
        }
    }
}

// thread per (edge,head): ex = exp(logit - max); accumulate denominator
__global__ void edge_exp(const int* __restrict__ dst, float* __restrict__ logits,
                         const float* __restrict__ rmax, float* __restrict__ rden, int E)
{
    long i = (long)blockIdx.x * 256 + threadIdx.x;
    if (i >= (long)E * 8) return;
    long edge = i >> 3; int h = (int)(i & 7);
    long col = dst[edge];
    float m = rmax[col * 8 + h];
    if (!(m > -1e37f && m < 1e37f)) m = 0.0f;   // mirror ref's isfinite guard
    float ex = expf(logits[i] - m);
    logits[i] = ex;
    atomicAdd(&rden[col * 8 + h], ex);
}

// wave-per-edge: agg[dst,h,:] += attn[e,h] * v[src,h,:]
__global__ __launch_bounds__(256) void edge_agg(
    const int* __restrict__ src, const int* __restrict__ dst,
    const bh16* __restrict__ Vf, const float* __restrict__ ex,
    const float* __restrict__ rden, float* __restrict__ agg, int E)
{
    long gid = (long)blockIdx.x * 256 + threadIdx.x;
    long edge = gid >> 5;
    int lane = (int)(gid & 31);
    if (edge >= E) return;
    long row = src[edge], col = dst[edge];
#pragma unroll
    for (int h = 0; h < 8; ++h) {
        float a = ex[edge * 8 + h] / (rden[col * 8 + h] + 1e-16f);
        float v = (float)Vf[row * 256 + h * 32 + lane] * a;
        atomicAdd(&agg[col * 256 + h * 32 + lane], v);
    }
}

__global__ void gelu_to_bf16(const float* __restrict__ in, bh16* __restrict__ out, long n)
{
    long i = (long)blockIdx.x * 256 + threadIdx.x;
    if (i >= n) return;
    float x = in[i];
    float g = 0.5f * x * (1.0f + erff(x * 0.7071067811865475f));
    out[i] = (bh16)g;
}

// x_new = sigmoid(skip[t]) * o + (1 - sigmoid(skip[t])) * x_old  (in place on XS)
__global__ void skip_combine(const bh16* __restrict__ O, bh16* __restrict__ XS,
                             const float* __restrict__ skipv, long n)
{
    long i = (long)blockIdx.x * 256 + threadIdx.x;
    if (i >= n) return;
    long node = i >> 8;
    int t = node < 50000 ? 0 : (node < 70000 ? 1 : 2);
    float g = 1.0f / (1.0f + expf(-skipv[t]));
    float v = g * (float)O[i] + (1.0f - g) * (float)XS[i];
    XS[i] = (bh16)v;
}

// ---------------------------------------------------------------------------
// Host side
// ---------------------------------------------------------------------------
static inline unsigned idiv(long a, long b) { return (unsigned)((a + b - 1) / b); }
static inline dim3 g1(long n) { return dim3(idiv(n, 256)); }

template <typename AT, typename CT>
static void launch_gemm(const AT* A, const bh16* Bt, const float* bias, CT* C,
                        int M, int N, int K, int act, hipStream_t s)
{
    dim3 grid(idiv(M, 128), (unsigned)(N / 128));
    gemm_wmma<AT, CT><<<grid, 256, 0, s>>>(A, Bt, bias, C, M, N, K, act);
}

extern "C" void kernel_launch(void* const* d_in, const int* in_sizes, int n_in,
                              void* d_out, int out_size, void* d_ws, size_t ws_size,
                              hipStream_t stream)
{
    const float* xin[3]  = {(const float*)d_in[0], (const float*)d_in[1], (const float*)d_in[2]};
    const int*   esrc[3] = {(const int*)d_in[3], (const int*)d_in[5], (const int*)d_in[7]};
    const int*   edst[3] = {(const int*)d_in[4], (const int*)d_in[6], (const int*)d_in[8]};
    const float *W1 = (const float*)d_in[9],  *b1 = (const float*)d_in[10];
    const float *W2 = (const float*)d_in[11], *b2 = (const float*)d_in[12];
    const float *Wk = (const float*)d_in[13], *bk = (const float*)d_in[14];
    const float *Wq = (const float*)d_in[15], *bq = (const float*)d_in[16];
    const float *Wv = (const float*)d_in[17], *bv = (const float*)d_in[18];
    const float *Wa = (const float*)d_in[19], *ba = (const float*)d_in[20];
    const float *skp   = (const float*)d_in[21];
    const float *a_rel = (const float*)d_in[22];
    const float *m_rel = (const float*)d_in[23];
    const float *p_rel = (const float*)d_in[24];
    float* out = (float*)d_out;

    const int  cnt[3]   = {50000, 20000, 200000};
    const long off[3]   = {0, 50000, 70000};
    const int  stype[3] = {1, 0, 2};   // src node type per edge type
    const int  dtype[3] = {2, 2, 0};   // dst node type per edge type
    const int  E = 200000;
    const long NT = 270000;
    const float invsqrtD = 0.17677669529663687f;   // 1/sqrt(32)

    // ---- carve workspace (all chunks multiples of 256B -> 16B alignment kept) ----
    size_t sAct = (size_t)NT * 256 * sizeof(bh16);      // 138.24 MB each
    size_t sAgg = (size_t)NT * 256 * sizeof(float);     // 276.48 MB
    size_t sLog = (size_t)E * 8 * sizeof(float);
    size_t sRed = (size_t)200000 * 8 * sizeof(float);
    size_t sW1t = (size_t)256 * 768 * sizeof(bh16);
    size_t sW2t = (size_t)128 * 256 * sizeof(bh16);
    size_t sWp  = (size_t)6 * 256 * 256 * sizeof(bh16); // WqT / WaT stacks
    size_t sWf  = (size_t)256 * 256 * sizeof(bh16);
    size_t sWfb = 1024;
    size_t need = 4 * sAct + sAgg + sLog + 2 * sRed + sW1t + sW2t + 2 * sWp + sWf + sWfb;
    if (ws_size < need) return;

    char* p = (char*)d_ws;
    bh16*  XS  = (bh16*)p;  p += sAct;   // activations (bf16)
    bh16*  KB  = (bh16*)p;  p += sAct;   // folded keys (reused as O buffer)
    bh16*  VB  = (bh16*)p;  p += sAct;   // folded values
    bh16*  QB  = (bh16*)p;  p += sAct;   // queries (reused as gelu(agg))
    float* AGG = (float*)p; p += sAgg;   // fp32 aggregation target for atomics
    float* LOG = (float*)p; p += sLog;   // logits -> exp weights
    float* RMX = (float*)p; p += sRed;
    float* RDN = (float*)p; p += sRed;
    bh16*  W1T = (bh16*)p;  p += sW1t;   // transposed bf16 weight panels
    bh16*  W2T = (bh16*)p;  p += sW2t;
    bh16*  WQT = (bh16*)p;  p += sWp;
    bh16*  WAT = (bh16*)p;  p += sWp;
    bh16*  WFT = (bh16*)p;  p += sWf;    // folded weight scratch (transposed bf16)
    float* WFb = (float*)p;

    // ---- pre-transpose all static weights to bf16 [N][K] panels ----
    transpose_to_bf16<<<g1(768 * 256), 256, 0, stream>>>(W1, W1T, 768, 256);
    transpose_to_bf16<<<g1(256 * 128), 256, 0, stream>>>(W2, W2T, 256, 128);
    for (int i = 0; i < 6; ++i) {
        transpose_to_bf16<<<g1(65536), 256, 0, stream>>>(Wq + (size_t)i * 65536,
                                                         WQT + (size_t)i * 65536, 256, 256);
        transpose_to_bf16<<<g1(65536), 256, 0, stream>>>(Wa + (size_t)i * 65536,
                                                         WAT + (size_t)i * 65536, 256, 256);
    }

    // ---- input MLP: XS = lrelu(x @ W1 + b1), bf16 ----
    for (int t = 0; t < 3; ++t)
        launch_gemm<float, bh16>(xin[t], W1T, b1, XS + off[t] * 256,
                                 cnt[t], 256, 768, 1, stream);

    for (int l = 0; l < 2; ++l) {
        // Q projections
        for (int t = 0; t < 3; ++t)
            launch_gemm<bh16, bh16>(XS + off[t] * 256,
                                    WQT + (size_t)(l * 3 + t) * 65536,
                                    bq + (size_t)(l * 3 + t) * 256,
                                    QB + off[t] * 256, cnt[t], 256, 256, 0, stream);
        // K/V with relation transform folded into (transposed bf16) weights
        for (int e = 0; e < 3; ++e) {
            int s = stype[e];
            fold_rel<<<dim3(257), dim3(256), 0, stream>>>(
                Wk + (size_t)(l * 3 + s) * 65536, bk + (size_t)(l * 3 + s) * 256,
                a_rel + (size_t)(l * 3 + e) * 8192, WFT, WFb);
            launch_gemm<bh16, bh16>(XS + off[s] * 256, WFT, WFb,
                                    KB + off[s] * 256, cnt[s], 256, 256, 0, stream);
            fold_rel<<<dim3(257), dim3(256), 0, stream>>>(
                Wv + (size_t)(l * 3 + s) * 65536, bv + (size_t)(l * 3 + s) * 256,
                m_rel + (size_t)(l * 3 + e) * 8192, WFT, WFb);
            launch_gemm<bh16, bh16>(XS + off[s] * 256, WFT, WFb,
                                    VB + off[s] * 256, cnt[s], 256, 256, 0, stream);
        }
        // attention + aggregation
        (void)hipMemsetAsync(AGG, 0, sAgg, stream);
        for (int e = 0; e < 3; ++e) {
            int s = stype[e], t = dtype[e];
            long nmh = (long)cnt[t] * 8;
            fill_kernel<<<g1(nmh), 256, 0, stream>>>(RMX, -INFINITY, nmh);
            fill_kernel<<<g1(nmh), 256, 0, stream>>>(RDN, 0.0f, nmh);
            edge_logits<<<g1((long)E * 32), 256, 0, stream>>>(
                esrc[e], edst[e], KB + off[s] * 256, QB + off[t] * 256,
                p_rel + (size_t)(l * 3 + e) * 8, LOG, RMX, E, invsqrtD);
            edge_exp<<<g1((long)E * 8), 256, 0, stream>>>(edst[e], LOG, RMX, RDN, E);
            edge_agg<<<g1((long)E * 32), 256, 0, stream>>>(
                esrc[e], edst[e], VB + off[s] * 256, LOG, RDN, AGG + off[t] * 256, E);
        }
        // GELU -> output projection -> gated skip
        gelu_to_bf16<<<g1(NT * 256), 256, 0, stream>>>(AGG, QB, NT * 256);
        for (int t = 0; t < 3; ++t)
            launch_gemm<bh16, bh16>(QB + off[t] * 256,
                                    WAT + (size_t)(l * 3 + t) * 65536,
                                    ba + (size_t)(l * 3 + t) * 256,
                                    KB + off[t] * 256, cnt[t], 256, 256, 0, stream);
        skip_combine<<<g1(NT * 256), 256, 0, stream>>>(KB, XS, skp + l * 3, NT * 256);
    }

    // ---- output MLP: out = lrelu(XS @ W2 + b2), fp32, concat over types ----
    for (int t = 0; t < 3; ++t)
        launch_gemm<bh16, float>(XS + off[t] * 256, W2T, b2,
                                 out + off[t] * 128, cnt[t], 128, 256, 1, stream);
}
